// Model_29119878266991
// MI455X (gfx1250) — matrix operand order repaired and checked by reading, not compiled, not run on hardware
//
#include <hip/hip_runtime.h>
#include <math.h>

// ---------------------------------------------------------------------------
// MI455X (gfx1250) implementation of the graph-transformer forward pass.
// All heavy GEMMs + attention use v_wmma_f32_16x16x32_f16 (wave32 WMMA),
// fp16 operands with fp32 accumulation. Edges are processed in chunks so the
// [2048,96,512] edge tensor / [.,2048] FFN hidden are never fully resident.
// GEMM: 128x128 block tile, double-buffered LDS (1 barrier / K-step),
// global_prefetch_b8 on the K+2 tile, 8 WMMAs per wave per K-step.
// ---------------------------------------------------------------------------

typedef __attribute__((ext_vector_type(16))) _Float16 v16h;
typedef __attribute__((ext_vector_type(8)))  float    v8f;
typedef _Float16 half_t;

constexpr int NN   = 128;    // nodes
constexpr int NE   = 2048;   // edges
constexpr int SQ   = 96;     // seq len
constexpr int DM   = 512;    // d_model
constexpr int DFF  = 2048;   // d_ff
constexpr int NHD  = 8;      // heads
constexpr int DHD  = 64;     // head dim
constexpr int CE   = 128;    // edges per chunk
constexpr int NCHUNK = NE / CE;          // 16
constexpr int MROWS  = CE * SQ;          // 12288 (== NN*SQ, shared buffers)
constexpr int PRED   = 48;

// ------------------------------ device helpers -----------------------------

__device__ __forceinline__ v8f vzero8() {
  v8f v;
#pragma unroll
  for (int i = 0; i < 8; ++i) v[i] = 0.f;
  return v;
}

__device__ __forceinline__ v8f wmma16(v16h a, v16h b, v8f c) {
  return __builtin_amdgcn_wmma_f32_16x16x32_f16(false, a, false, b,
                                                (short)0, c, false, false);
}

// A-fragment (16x32, M x K) per CDNA5 ISA layout:
//  lane l: row = l&15, kb = (l>>4)*8 ; f[i]=K(kb+i), f[8+i]=K(16+kb+i)
__device__ __forceinline__ v16h load_fragA(const half_t* base, int stride) {
  const int lane = threadIdx.x & 31;
  const int row  = lane & 15;
  const int kb   = (lane >> 4) * 8;
  const half_t* p = base + row * stride;
  v16h f;
#pragma unroll
  for (int i = 0; i < 8; ++i) { f[i] = p[kb + i]; f[8 + i] = p[16 + kb + i]; }
  return f;
}

// B-fragment (32x16, K x N), staged transposed (base[n*stride + k]):
//  lane l: col n = l&15, kb = (l>>4)*8 ; f[i]=K(kb+i), f[8+i]=K(16+kb+i)
__device__ __forceinline__ v16h load_fragB(const half_t* base, int stride) {
  const int lane = threadIdx.x & 31;
  const int n    = lane & 15;
  const int kb   = (lane >> 4) * 8;
  const half_t* p = base + n * stride + kb;
  v16h f;
#pragma unroll
  for (int i = 0; i < 8; ++i) { f[i] = p[i]; f[8 + i] = p[16 + i]; }
  return f;
}

__device__ __forceinline__ float gelu_f(float x) {
  const float c = 0.7978845608028654f;     // sqrt(2/pi)
  float t = tanhf(c * (x + 0.044715f * x * x * x));
  return 0.5f * x * (1.f + t);
}

// ------------------------------ small kernels ------------------------------

__global__ void f32_to_f16_kernel(const float* __restrict__ in,
                                  half_t* __restrict__ out, int n) {
  int i = blockIdx.x * blockDim.x + threadIdx.x;
  if (i < n) out[i] = (half_t)in[i];
}

__global__ void zero_f32_kernel(float* __restrict__ p, int n) {
  int i = blockIdx.x * blockDim.x + threadIdx.x;
  if (i < n) p[i] = 0.f;
}

// nodes = conv1d(x_dec, W_token, circular) + x_mark @ W_time + PE
__global__ void embed_nodes_kernel(const float* __restrict__ xdec,
                                   const float* __restrict__ xmark,
                                   const float* __restrict__ Wtok,
                                   const float* __restrict__ Wtime,
                                   float* __restrict__ nodesF,
                                   half_t* __restrict__ nodesH) {
  int idx = blockIdx.x * blockDim.x + threadIdx.x;
  if (idx >= NN * SQ * DM) return;
  int d = idx & (DM - 1);
  int s = (idx >> 9) % SQ;
  int n = idx / (SQ * DM);
  float acc = 0.f;
#pragma unroll
  for (int k = 0; k < 3; ++k) {
    int ss = (s - 1 + k + SQ) % SQ;
    const float* xp = xdec + ((size_t)n * SQ + ss) * 7;
    const float* wp = Wtok + (size_t)d * 21 + k;   // W[d][i][k]
#pragma unroll
    for (int i = 0; i < 7; ++i) acc += xp[i] * wp[i * 3];
  }
  const float* mp = xmark + ((size_t)n * SQ + s) * 4;
#pragma unroll
  for (int m = 0; m < 4; ++m) acc += mp[m] * Wtime[m * DM + d];
  int j = d >> 1;
  float div = expf((float)(2 * j) * (-9.210340371976184f / (float)DM));
  float ang = (float)s * div;
  acc += (d & 1) ? cosf(ang) : sinf(ang);
  nodesF[idx] = acc;
  nodesH[idx] = (half_t)acc;
}

// e0[e,d] = edge_attr[e]*W_eemb[d] + b_eemb[d]
__global__ void embed_edges0_kernel(const float* __restrict__ eattr,
                                    const float* __restrict__ We,
                                    const float* __restrict__ be,
                                    float* __restrict__ e0f,
                                    half_t* __restrict__ e0h) {
  int idx = blockIdx.x * blockDim.x + threadIdx.x;
  if (idx >= NE * DM) return;
  int d = idx & (DM - 1);
  int e = idx >> 9;
  float v = eattr[e] * We[d] + be[d];
  e0f[idx] = v;
  e0h[idx] = (half_t)v;
}

__global__ void count_recv_kernel(const int* __restrict__ recv,
                                  float* __restrict__ cnt) {
  int e = blockIdx.x * blockDim.x + threadIdx.x;
  if (e < NE) atomicAdd(&cnt[recv[e]], 1.0f);
}

// cat[e_local*96+s][0:512]=e0[e], [512:1024]=nodes[snd], [1024:1536]=nodes[rcv]
__global__ void gather_cat_kernel(const half_t* __restrict__ e0h,
                                  const half_t* __restrict__ nodesH,
                                  const int* __restrict__ snd,
                                  const int* __restrict__ rcv,
                                  half_t* __restrict__ cat, int e_base) {
  int idx = blockIdx.x * blockDim.x + threadIdx.x;
  if (idx >= CE * SQ * DM) return;
  int d = idx & (DM - 1);
  int s = (idx >> 9) % SQ;
  int el = idx / (SQ * DM);
  int e = e_base + el;
  size_t row = ((size_t)el * SQ + s) * (3 * DM);
  cat[row + d]            = e0h[(size_t)e * DM + d];
  cat[row + DM + d]       = nodesH[((size_t)snd[e] * SQ + s) * DM + d];
  cat[row + 2 * DM + d]   = nodesH[((size_t)rcv[e] * SQ + s) * DM + d];
}

// agg[rcv[e],s,d] += encOut[el,s,d] + e0[e,d]
__global__ void scatter_edges_kernel(const float* __restrict__ encOut,
                                     const float* __restrict__ e0f,
                                     const int* __restrict__ rcv,
                                     float* __restrict__ agg, int e_base) {
  int idx = blockIdx.x * blockDim.x + threadIdx.x;
  if (idx >= CE * SQ * DM) return;
  int d = idx & (DM - 1);
  int s = (idx >> 9) % SQ;
  int el = idx / (SQ * DM);
  int e = e_base + el;
  float v = encOut[idx] + e0f[(size_t)e * DM + d];
  atomicAdd(&agg[((size_t)rcv[e] * SQ + s) * DM + d], v);
}

__global__ void mean_agg_kernel(float* __restrict__ agg,
                                const float* __restrict__ cnt,
                                half_t* __restrict__ aggH) {
  int idx = blockIdx.x * blockDim.x + threadIdx.x;
  if (idx >= NN * SQ * DM) return;
  int n = idx / (SQ * DM);
  float v = agg[idx] / fmaxf(cnt[n], 1.0f);
  agg[idx] = v;
  aggH[idx] = (half_t)v;
}

__global__ void cat_node_kernel(const half_t* __restrict__ nodesH,
                                const half_t* __restrict__ aggH,
                                half_t* __restrict__ cat2) {
  int idx = blockIdx.x * blockDim.x + threadIdx.x;
  if (idx >= NN * SQ * DM) return;
  int d = idx & (DM - 1);
  size_t row = (size_t)(idx / DM) * (2 * DM);
  cat2[row + d]      = nodesH[idx];
  cat2[row + DM + d] = aggH[idx];
}

// out = LN(X + Y) * g + b  over rows of 512; optional f16 copy
__global__ void __launch_bounds__(256)
add_ln_kernel(const float* __restrict__ X, const float* __restrict__ Y,
              const float* __restrict__ g, const float* __restrict__ be,
              float* __restrict__ Of, half_t* __restrict__ Oh) {
  const int row = blockIdx.x, tid = threadIdx.x;
  const size_t base = (size_t)row * DM;
  float a0 = X[base + tid] + Y[base + tid];
  float a1 = X[base + tid + 256] + Y[base + tid + 256];
  __shared__ float r1[256], r2[256];
  r1[tid] = a0 + a1;
  r2[tid] = a0 * a0 + a1 * a1;
  __syncthreads();
  for (int s = 128; s > 0; s >>= 1) {
    if (tid < s) { r1[tid] += r1[tid + s]; r2[tid] += r2[tid + s]; }
    __syncthreads();
  }
  float mean = r1[0] * (1.f / DM);
  float var  = r2[0] * (1.f / DM) - mean * mean;
  float inv  = rsqrtf(var + 1e-5f);
  float v0 = (a0 - mean) * inv * g[tid] + be[tid];
  float v1 = (a1 - mean) * inv * g[tid + 256] + be[tid + 256];
  Of[base + tid] = v0;
  Of[base + tid + 256] = v1;
  if (Oh) { Oh[base + tid] = (half_t)v0; Oh[base + tid + 256] = (half_t)v1; }
}

// out[n,sr] = sum_d (encOut[n,48+sr,d] + nodes[n,48+sr,d]) * projW[d] + projb
__global__ void __launch_bounds__(128)
proj_kernel(const float* __restrict__ encOut, const float* __restrict__ nodesF,
            const float* __restrict__ projW, const float* __restrict__ projb,
            float* __restrict__ out) {
  int row = blockIdx.x;                 // n*48 + sr
  int n = row / PRED;
  int s = (SQ - PRED) + (row % PRED);
  const float* a = encOut + ((size_t)n * SQ + s) * DM;
  const float* b = nodesF + ((size_t)n * SQ + s) * DM;
  float sum = 0.f;
  for (int d = threadIdx.x; d < DM; d += 128) sum += (a[d] + b[d]) * projW[d];
  __shared__ float red[128];
  red[threadIdx.x] = sum;
  __syncthreads();
  for (int st = 64; st > 0; st >>= 1) {
    if (threadIdx.x < st) red[threadIdx.x] += red[threadIdx.x + st];
    __syncthreads();
  }
  if (threadIdx.x == 0) out[row] = red[0] + projb[0];
}

// ------------------------------ WMMA GEMM ----------------------------------
// C[M,N] = act(A[M,K] @ B[K,N] + bias);  A,B f16 row-major; out f32/f16.
// Block tile 128x128, 8 waves (4 M x 2 N), each wave 32x64 = 2x4 WMMA accs.
// Double-buffered LDS: one barrier per K-step; next tile loaded to registers
// while WMMAs run; global_prefetch_b8 issued for the K+2 tile.
#define BM 128
#define BN 128
#define BK 32
#define LDA 40   // halves: 32 + 8 pad
#define LDB 40   // Bs transposed [n][k]

__global__ void __launch_bounds__(256)
gemm_f16_kernel(const half_t* __restrict__ A, const half_t* __restrict__ B,
                const float* __restrict__ bias, float* __restrict__ Cf,
                half_t* __restrict__ Ch, int M, int N, int K, int act) {
  __shared__ __align__(16) half_t As[2][BM * LDA];   // 2*10240 B
  __shared__ __align__(16) half_t Bs[2][BN * LDB];   // 2*10240 B
  const int tid  = threadIdx.x;
  const int wave = tid >> 5;
  const int m0 = blockIdx.y * BM;
  const int n0 = blockIdx.x * BN;
  const int wm = (wave >> 1) * 32;
  const int wn = (wave & 1) * 64;

  // per-thread staging coords
  const int arow = tid >> 1, acol = (tid & 1) * 16;   // A: 128 rows x 32
  const int brow = tid >> 3, bcol = (tid & 7) * 16;   // B: 32 rows x 128
  const half_t* gA = A + (size_t)(m0 + arow) * K + acol;
  const half_t* gB = B + (size_t)brow * N + n0 + bcol;

  v8f acc[2][4];
#pragma unroll
  for (int i = 0; i < 2; ++i)
#pragma unroll
    for (int j = 0; j < 4; ++j) acc[i][j] = vzero8();

  const int NT = K / BK;
  uint4 ra0, ra1, rb0, rb1;

  // tile 0 -> registers -> LDS buffer 0
  ra0 = *(const uint4*)(gA);
  ra1 = *(const uint4*)(gA + 8);
  rb0 = *(const uint4*)(gB);
  rb1 = *(const uint4*)(gB + 8);
  {
    half_t* sa = &As[0][arow * LDA + acol];
    *(uint4*)(sa)     = ra0;
    *(uint4*)(sa + 8) = ra1;
    half_t tmp[16];
    *(uint4*)(tmp)     = rb0;
    *(uint4*)(tmp + 8) = rb1;
#pragma unroll
    for (int i = 0; i < 16; ++i) Bs[0][(bcol + i) * LDB + brow] = tmp[i];
  }
  __syncthreads();

  for (int kt = 0; kt < NT; ++kt) {
    const int cur = kt & 1;
    if (kt + 1 < NT) {       // issue next-tile global loads before compute
      const half_t* pa = gA + (size_t)(kt + 1) * BK;
      const half_t* pb = gB + (size_t)(kt + 1) * BK * N;
      ra0 = *(const uint4*)(pa);
      ra1 = *(const uint4*)(pa + 8);
      rb0 = *(const uint4*)(pb);
      rb1 = *(const uint4*)(pb + 8);
      if (kt + 2 < NT) {     // global_prefetch_b8 for the K+2 tile
        __builtin_prefetch(gA + (size_t)(kt + 2) * BK, 0, 3);
        __builtin_prefetch(gB + (size_t)(kt + 2) * BK * N, 0, 3);
      }
    }

    v16h af[2], bf[4];
    af[0] = load_fragA(&As[cur][wm * LDA], LDA);
    af[1] = load_fragA(&As[cur][(wm + 16) * LDA], LDA);
#pragma unroll
    for (int j = 0; j < 4; ++j)
      bf[j] = load_fragB(&Bs[cur][(wn + j * 16) * LDB], LDB);
#pragma unroll
    for (int i = 0; i < 2; ++i)
#pragma unroll
      for (int j = 0; j < 4; ++j) acc[i][j] = wmma16(af[i], bf[j], acc[i][j]);

    if (kt + 1 < NT) {       // stage next tile into the other buffer
      const int nxt = cur ^ 1;
      half_t* sa = &As[nxt][arow * LDA + acol];
      *(uint4*)(sa)     = ra0;
      *(uint4*)(sa + 8) = ra1;
      half_t tmp[16];
      *(uint4*)(tmp)     = rb0;
      *(uint4*)(tmp + 8) = rb1;
#pragma unroll
      for (int i = 0; i < 16; ++i) Bs[nxt][(bcol + i) * LDB + brow] = tmp[i];
    }
    __syncthreads();
  }

  // epilogue: D layout -> lane (l>>4)*8+p rows, (l&15) cols per 16x16 tile
  const int lane  = tid & 31;
  const int rbase = (lane >> 4) * 8;
  const int c     = lane & 15;
#pragma unroll
  for (int i = 0; i < 2; ++i)
#pragma unroll
    for (int j = 0; j < 4; ++j) {
      int gn = n0 + wn + j * 16 + c;
      float bv = bias ? bias[gn] : 0.f;
#pragma unroll
      for (int p = 0; p < 8; ++p) {
        int gm = m0 + wm + i * 16 + rbase + p;
        float v = acc[i][j][p] + bv;
        if (act == 1) v = gelu_f(v);
        size_t o = (size_t)gm * N + gn;
        if (Cf) Cf[o] = v;
        if (Ch) Ch[o] = (half_t)v;
      }
    }
}

// ------------------------------ WMMA attention -----------------------------
// One block per (batch, head). S=96, dh=64. Q,K staged row-major; V staged
// transposed; scores->softmax in LDS (f16 probs); both MMA stages use WMMA.
#define LDQ 72
#define LDV 104
#define LDP 104

__global__ void __launch_bounds__(256)
attn_kernel(const half_t* __restrict__ Q, const half_t* __restrict__ Km,
            const half_t* __restrict__ V, half_t* __restrict__ O) {
  __shared__ __align__(16) half_t Qs[SQ * LDQ];
  __shared__ __align__(16) half_t Ks[SQ * LDQ];
  __shared__ __align__(16) half_t Vt[DHD * LDV];
  __shared__ __align__(16) half_t Pf[SQ * LDP];

  const int tid = threadIdx.x;
  const int wave = tid >> 5;
  const int b = blockIdx.x / NHD;
  const int h = blockIdx.x % NHD;
  const size_t gbase = ((size_t)b * SQ) * DM + h * DHD;

  // stage Q, K (96x64 each, 8-half vectors) and V transposed
  for (int idx = tid; idx < SQ * 8; idx += 256) {
    int row = idx >> 3, c0 = (idx & 7) * 8;
    *(uint4*)&Qs[row * LDQ + c0] = *(const uint4*)(Q + gbase + (size_t)row * DM + c0);
    *(uint4*)&Ks[row * LDQ + c0] = *(const uint4*)(Km + gbase + (size_t)row * DM + c0);
    half_t tmp[8];
    *(uint4*)tmp = *(const uint4*)(V + gbase + (size_t)row * DM + c0);
#pragma unroll
    for (int i = 0; i < 8; ++i) Vt[(c0 + i) * LDV + row] = tmp[i];
  }
  __syncthreads();

  const int lane = tid & 31;
  const int rbase = (lane >> 4) * 8;
  const int cc = lane & 15;

  // scores: 6x6 tiles of 16x16, contraction over dh (2 k-steps of 32)
  for (int tile = wave; tile < 36; tile += 8) {
    int ti = tile / 6, tj = tile % 6;
    v8f acc = vzero8();
#pragma unroll
    for (int d0 = 0; d0 < DHD; d0 += 32) {
      v16h a = load_fragA(Qs + (ti * 16) * LDQ + d0, LDQ);
      v16h bfr = load_fragB(Ks + (tj * 16) * LDQ + d0, LDQ);  // B[k=d][n=t]
      acc = wmma16(a, bfr, acc);
    }
#pragma unroll
    for (int p = 0; p < 8; ++p) {
      int r = ti * 16 + rbase + p, c = tj * 16 + cc;
      Pf[r * LDP + c] = (half_t)(acc[p] * 0.125f);   // 1/sqrt(64)
    }
  }
  __syncthreads();

  // softmax per row (one thread per row)
  if (tid < SQ) {
    int r = tid;
    float m = -1e30f;
    for (int j = 0; j < SQ; ++j) m = fmaxf(m, (float)Pf[r * LDP + j]);
    float ssum = 0.f;
    for (int j = 0; j < SQ; ++j) {
      float e = expf((float)Pf[r * LDP + j] - m);
      ssum += e;
      Pf[r * LDP + j] = (half_t)e;
    }
    float inv = 1.f / ssum;
    for (int j = 0; j < SQ; ++j)
      Pf[r * LDP + j] = (half_t)((float)Pf[r * LDP + j] * inv);
  }
  __syncthreads();

  // O = P @ V : 6x4 tiles, contraction over t (3 k-steps of 32)
  for (int tile = wave; tile < 24; tile += 8) {
    int ti = tile / 4, tj = tile % 4;
    v8f acc = vzero8();
#pragma unroll
    for (int t0 = 0; t0 < SQ; t0 += 32) {
      v16h a = load_fragA(Pf + (ti * 16) * LDP + t0, LDP);
      v16h bfr = load_fragB(Vt + (tj * 16) * LDV + t0, LDV);  // B[k=t][n=d]
      acc = wmma16(a, bfr, acc);
    }
#pragma unroll
    for (int p = 0; p < 8; ++p) {
      int s = ti * 16 + rbase + p;
      int d = h * DHD + tj * 16 + cc;
      O[((size_t)b * SQ + s) * DM + d] = (half_t)acc[p];
    }
  }
}

// ------------------------------ host orchestration -------------------------

struct EncWeights {
  const half_t *Wq, *Wk, *Wv, *Wo, *W1, *W2;
  const float *bq, *bk, *bv, *bo, *b1, *b2, *g1, *be1, *g2, *be2;
};

extern "C" void kernel_launch(void* const* d_in, const int* in_sizes, int n_in,
                              void* d_out, int out_size, void* d_ws, size_t ws_size,
                              hipStream_t stream) {
  (void)in_sizes; (void)n_in; (void)out_size; (void)ws_size;

  // ---- inputs (setup_inputs insertion order; params flattened in order) ----
  const float* x_dec      = (const float*)d_in[2];
  const float* x_mark_dec = (const float*)d_in[3];
  const float* edge_attr  = (const float*)d_in[4];
  const int*   senders    = (const int*)d_in[5];
  const int*   receivers  = (const int*)d_in[6];
  const float* W_token    = (const float*)d_in[7];
  const float* W_time     = (const float*)d_in[8];
  const float* W_eemb     = (const float*)d_in[9];
  const float* b_eemb     = (const float*)d_in[10];
  const float* edge_fc_W  = (const float*)d_in[11];
  const float* edge_fc_b  = (const float*)d_in[12];
  const float* node_fc_W  = (const float*)d_in[13];
  const float* node_fc_b  = (const float*)d_in[14];
  const float* eW[16]; for (int i = 0; i < 16; ++i) eW[i] = (const float*)d_in[15 + i];
  const float* nW[16]; for (int i = 0; i < 16; ++i) nW[i] = (const float*)d_in[31 + i];
  const float* proj_W = (const float*)d_in[47];
  const float* proj_b = (const float*)d_in[48];
  float* out = (float*)d_out;

  // ---- workspace layout ----
  char* wsb = (char*)d_ws;
  size_t off = 0;
  auto alloc = [&](size_t bytes) -> void* {
    void* p = wsb + off;
    off += (bytes + 255) & ~(size_t)255;
    return p;
  };
  const size_t ND = (size_t)NN * SQ * DM;      // 6.29M
  float*  nodesF = (float*)alloc(ND * 4);
  half_t* nodesH = (half_t*)alloc(ND * 2);
  float*  e0F    = (float*)alloc((size_t)NE * DM * 4);
  half_t* e0H    = (half_t*)alloc((size_t)NE * DM * 2);
  float*  aggF   = (float*)alloc(ND * 4);
  half_t* aggH   = (half_t*)alloc(ND * 2);
  float*  cnt    = (float*)alloc(NN * 4);
  half_t* edgefcWh = (half_t*)alloc((size_t)3 * DM * DM * 2);
  half_t* nodefcWh = (half_t*)alloc((size_t)2 * DM * DM * 2);
  half_t* encWh[2][6];   // Wq Wk Wv Wo W1 W2 for edge/node enc
  const size_t wsz[6] = {(size_t)DM*DM,(size_t)DM*DM,(size_t)DM*DM,(size_t)DM*DM,
                         (size_t)DM*DFF,(size_t)DFF*DM};
  for (int e = 0; e < 2; ++e)
    for (int i = 0; i < 6; ++i) encWh[e][i] = (half_t*)alloc(wsz[i] * 2);
  const size_t MD = (size_t)MROWS * DM;        // 6.29M
  half_t* catH = (half_t*)alloc((size_t)MROWS * 3 * DM * 2);
  float*  xF   = (float*)alloc(MD * 4);
  half_t* xH   = (half_t*)alloc(MD * 2);
  half_t* qH   = (half_t*)alloc(MD * 2);
  half_t* kH   = (half_t*)alloc(MD * 2);
  half_t* vH   = (half_t*)alloc(MD * 2);
  half_t* oH   = (half_t*)alloc(MD * 2);
  float*  t1F  = (float*)alloc(MD * 4);
  float*  x2F  = (float*)alloc(MD * 4);
  half_t* x2H  = (half_t*)alloc(MD * 2);
  half_t* hH   = (half_t*)alloc((size_t)MROWS * DFF * 2);

  auto conv = [&](const float* s, half_t* d, int n) {
    f32_to_f16_kernel<<<(n + 255) / 256, 256, 0, stream>>>(s, d, n);
  };
  auto gemm = [&](const half_t* A, const half_t* B, const float* bias,
                  float* Cf, half_t* Ch, int M, int N, int K, int act) {
    dim3 g(N / BN, M / BM);
    gemm_f16_kernel<<<g, 256, 0, stream>>>(A, B, bias, Cf, Ch, M, N, K, act);
  };

  // ---- weight conversion to f16 ----
  conv(edge_fc_W, edgefcWh, 3 * DM * DM);
  conv(node_fc_W, nodefcWh, 2 * DM * DM);
  const int wIdx[6] = {0, 2, 4, 6, 8, 10};   // Wq Wk Wv Wo W1 W2 positions
  for (int i = 0; i < 6; ++i) { conv(eW[wIdx[i]], encWh[0][i], (int)wsz[i]);
                                conv(nW[wIdx[i]], encWh[1][i], (int)wsz[i]); }

  EncWeights edgeEW = {encWh[0][0],encWh[0][1],encWh[0][2],encWh[0][3],encWh[0][4],encWh[0][5],
                       eW[1],eW[3],eW[5],eW[7],eW[9],eW[11],eW[12],eW[13],eW[14],eW[15]};
  EncWeights nodeEW = {encWh[1][0],encWh[1][1],encWh[1][2],encWh[1][3],encWh[1][4],encWh[1][5],
                       nW[1],nW[3],nW[5],nW[7],nW[9],nW[11],nW[12],nW[13],nW[14],nW[15]};

  auto run_encoder = [&](const EncWeights& ew) {  // x in xF/xH, out -> xF
    gemm(xH, ew.Wq, ew.bq, nullptr, qH, MROWS, DM, DM, 0);
    gemm(xH, ew.Wk, ew.bk, nullptr, kH, MROWS, DM, DM, 0);
    gemm(xH, ew.Wv, ew.bv, nullptr, vH, MROWS, DM, DM, 0);
    attn_kernel<<<(MROWS / SQ) * NHD, 256, 0, stream>>>(qH, kH, vH, oH);
    gemm(oH, ew.Wo, ew.bo, t1F, nullptr, MROWS, DM, DM, 0);
    add_ln_kernel<<<MROWS, 256, 0, stream>>>(xF, t1F, ew.g1, ew.be1, x2F, x2H);
    gemm(x2H, ew.W1, ew.b1, nullptr, hH, MROWS, DFF, DM, 1);   // fused GELU
    gemm(hH, ew.W2, ew.b2, t1F, nullptr, MROWS, DM, DFF, 0);
    add_ln_kernel<<<MROWS, 256, 0, stream>>>(x2F, t1F, ew.g2, ew.be2, xF, nullptr);
  };

  // ---- embeddings ----
  embed_nodes_kernel<<<(int)(ND / 256), 256, 0, stream>>>(
      x_dec, x_mark_dec, W_token, W_time, nodesF, nodesH);
  embed_edges0_kernel<<<(NE * DM) / 256, 256, 0, stream>>>(
      edge_attr, W_eemb, b_eemb, e0F, e0H);
  zero_f32_kernel<<<(int)(ND / 256), 256, 0, stream>>>(aggF, (int)ND);
  zero_f32_kernel<<<1, 256, 0, stream>>>(cnt, NN);
  count_recv_kernel<<<NE / 256, 256, 0, stream>>>(receivers, cnt);

  // ---- edge block, streamed in chunks ----
  for (int c = 0; c < NCHUNK; ++c) {
    int e_base = c * CE;
    gather_cat_kernel<<<(int)(MD / 256), 256, 0, stream>>>(
        e0H, nodesH, senders, receivers, catH, e_base);
    gemm(catH, edgefcWh, edge_fc_b, xF, xH, MROWS, DM, 3 * DM, 0);  // einp
    run_encoder(edgeEW);
    scatter_edges_kernel<<<(int)(MD / 256), 256, 0, stream>>>(
        xF, e0F, receivers, aggF, e_base);
  }

  // ---- node block ----
  mean_agg_kernel<<<(int)(ND / 256), 256, 0, stream>>>(aggF, cnt, aggH);
  cat_node_kernel<<<(int)(ND / 256), 256, 0, stream>>>(nodesH, aggH, catH);
  gemm(catH, nodefcWh, node_fc_b, xF, xH, MROWS, DM, 2 * DM, 0);    // ninp
  run_encoder(nodeEW);

  // ---- projection + slice ----
  proj_kernel<<<NN * PRED, 128, 0, stream>>>(xF, nodesF, proj_W, proj_b, out);
}
